// GradientLoss_15333033247161
// MI455X (gfx1250) — matrix operand order repaired and checked by reading, not compiled, not run on hardware
//
#include <hip/hip_runtime.h>

typedef __attribute__((ext_vector_type(16))) _Float16 v16h;
typedef __attribute__((ext_vector_type(8)))  float    v8f;

#define NB        2048          // fine histogram bins
#define NA        4             // odx, ody, tdx, tdy
#define NROWS     5             // NA rows + one zero row for WMMA padding
#define IMG_W     512
#define IMG_PIX   (512*512)
#define GH        511
#define GW        511
#define N_IMG     48
#define TOTAL_F   ((float)(N_IMG) * (float)(GH) * (float)(GW))
#define HRANGE    2.0f
#define HSCALE    512.0f        // NB / (2*HRANGE)
#define S_COEF    (-25.0f)      // -1/DELTA^2, DELTA=0.2

// ---------------- pass 0: zero workspace ----------------
__global__ void init_ws(unsigned int* ws) {
    int i = blockIdx.x * blockDim.x + threadIdx.x;
    if (i < NA * NB + 1) ws[i] = 0u;
}

// ---------------- pass 1: gradients, loss, fine histogram ----------------
__global__ __launch_bounds__(256) void grad_hist_kernel(
    const float* __restrict__ outp, const float* __restrict__ tgtp,
    unsigned int* __restrict__ ghist, float* __restrict__ lossAcc)
{
    __shared__ unsigned int lhist[NA * NB];   // 32 KB
    __shared__ float red[256];

    for (int i = threadIdx.x; i < NA * NB; i += blockDim.x) lhist[i] = 0u;
    __syncthreads();

    const int img = blockIdx.y;
    const float* ob = outp + (size_t)img * IMG_PIX;
    const float* tb = tgtp + (size_t)img * IMG_PIX;

    float lsum = 0.0f;
    for (int h = blockIdx.x; h < GH; h += gridDim.x) {
        const int rbase = h * IMG_W;
        for (int w = threadIdx.x; w < GW; w += blockDim.x) {
            const int base = rbase + w;
            float o00 = ob[base], o10 = ob[base + IMG_W], o01 = ob[base + 1];
            float t00 = tb[base], t10 = tb[base + IMG_W], t01 = tb[base + 1];
            float odx = o10 - o00, ody = o01 - o00;
            float tdx = t10 - t00, tdy = t01 - t00;
            float ex = odx - tdx, ey = ody - tdy;
            lsum += ex * ex + ey * ey;
            float g4[4] = {odx, ody, tdx, tdy};
#pragma unroll
            for (int a = 0; a < 4; ++a) {
                int b = (int)floorf((g4[a] + HRANGE) * HSCALE);
                // |g| > 2 has soft-kernel weight <= e^-81: drop (also keeps
                // counts small enough for near-exact f16 representation)
                if ((unsigned)b < (unsigned)NB)
                    atomicAdd(&lhist[a * NB + b], 1u);
            }
        }
    }
    __syncthreads();

    // merge per-block histogram into global (skip zeros)
    for (int i = threadIdx.x; i < NA * NB; i += blockDim.x) {
        unsigned int c = lhist[i];
        if (c) atomicAdd(&ghist[i], c);
    }

    // block loss reduction
    red[threadIdx.x] = lsum;
    __syncthreads();
    for (int s = 128; s > 0; s >>= 1) {
        if (threadIdx.x < s) red[threadIdx.x] += red[threadIdx.x + s];
        __syncthreads();
    }
    if (threadIdx.x == 0) atomicAdd(lossAcc, red[0]);
}

// ---------------- pass 2: soft histogram = counts x weights via WMMA ----------------
// hist[a][j] = (1/N) * sum_k cnt[a][k] * exp(-25*(c_k - m_j)^2)
// One workgroup, 4 wave32s; each wave computes one 16-column tile of the
// 16x64 (rows 0..3 live) result with v_wmma_f32_16x16x32_f16, K = 2048.
// Counts are staged in a 5-row LDS table whose last row is all zeros; lanes
// with A-row >= 4 read that zero row (clamped index) so all LDS loads are
// unconditional -> no EXEC predication around the WMMA loop.
__global__ __launch_bounds__(128) void hist_wmma_kernel(
    const unsigned int* __restrict__ ghist, const float* __restrict__ lossAcc,
    float* __restrict__ outv)
{
    __shared__ _Float16 lh[NROWS * NB];       // 20 KB
    for (int i = threadIdx.x; i < NA * NB; i += blockDim.x)
        lh[i] = (_Float16)(float)ghist[i];
    for (int i = NA * NB + threadIdx.x; i < NROWS * NB; i += blockDim.x)
        lh[i] = (_Float16)0.0f;               // zero pad row
    __syncthreads();

    const int lane = threadIdx.x & 31;
    const int wave = threadIdx.x >> 5;        // N-tile: bins [wave*16, wave*16+16)
    const int n    = lane & 15;               // column within tile / A-row per lane
    const int hi   = (lane >> 4) & 1;         // lane half selects K sub-range
    const int m    = (n < NA) ? n : NA;       // clamped A-row -> zero row if >= 4
    const int j    = wave * 16 + n;           // output bin index
    const float mj = -0.2f + ((float)j + 0.5f) * (0.4f / 64.0f);
    const _Float16* arow = lh + m * NB;

    v8f acc = {};
    for (int kb = 0; kb < NB; kb += 32) {
        v16h A, B;
#pragma unroll
        for (int v = 0; v < 8; ++v) {
            // A (16x32 f16): lanes0-15 M=lane, VGPR0-3 -> K 0..7, VGPR4-7 -> K 16..23
            //                lanes16-31 M=lane-16, VGPR0-3 -> K 8..15, VGPR4-7 -> K 24..31
            int ka = (v < 4) ? (hi * 8 + 2 * v) : (16 + hi * 8 + 2 * (v - 4));
            A[2 * v]     = arow[kb + ka];
            A[2 * v + 1] = arow[kb + ka + 1];

            // B (32x16 f16): col = lane%16; same K order as A (lanes0-15 K 0..7,16..23; lanes16-31 K 8..15,24..31)
            int kbB = (v < 4) ? (hi * 8 + 2 * v) : (16 + hi * 8 + 2 * (v - 4));
            float c0 = -HRANGE + ((float)(kb + kbB) + 0.5f) * (1.0f / HSCALE);
            float c1 = c0 + (1.0f / HSCALE);
            float d0 = c0 - mj, d1 = c1 - mj;
            B[2 * v]     = (_Float16)__expf(S_COEF * d0 * d0);
            B[2 * v + 1] = (_Float16)__expf(S_COEF * d1 * d1);
        }
        acc = __builtin_amdgcn_wmma_f32_16x16x32_f16(
            /*neg_a=*/false, A, /*neg_b=*/false, B,
            /*c_mod=*/(short)0, acc, /*reuse_a=*/false, /*reuse_b=*/false);
    }

    const float invN = 1.0f / TOTAL_F;
    // C/D f32 16x16 layout: lanes 0-15 col N=lane, VGPR r holds row M=r.
    // Rows 0..3 (odx,ody,tdx,tdy) live in VGPRs 0..3 of lanes 0..15.
    if (hi == 0) {
#pragma unroll
        for (int a = 0; a < 4; ++a)
            outv[1 + a * 64 + j] = acc[a] * invN;
    }
    if (threadIdx.x == 0) outv[0] = lossAcc[0] * invN;
}

// ---------------- launcher ----------------
extern "C" void kernel_launch(void* const* d_in, const int* in_sizes, int n_in,
                              void* d_out, int out_size, void* d_ws, size_t ws_size,
                              hipStream_t stream) {
    const float* outp = (const float*)d_in[0];
    const float* tgtp = (const float*)d_in[1];
    unsigned int* ws    = (unsigned int*)d_ws;
    unsigned int* ghist = ws;                     // NA*NB u32
    float*        lossA = (float*)(ws + NA * NB); // 1 f32
    float*        dout  = (float*)d_out;          // 257 f32

    init_ws<<<(NA * NB + 1 + 255) / 256, 256, 0, stream>>>(ws);

    dim3 grid(32, N_IMG);                         // 1536 blocks total
    grad_hist_kernel<<<grid, 256, 0, stream>>>(outp, tgtp, ghist, lossA);

    hist_wmma_kernel<<<1, 128, 0, stream>>>(ghist, lossA, dout);
}
